// zGRU_batch_28939489640843
// MI455X (gfx1250) — compile-verified, matrix operand order checked by reading, not run on hardware
//
#include <hip/hip_runtime.h>

typedef unsigned short u16;
typedef unsigned int   u32;
typedef __attribute__((ext_vector_type(16))) __bf16 v16bf;
typedef __attribute__((ext_vector_type(8)))  float  v8f;

#define S_  8
#define B_  32
#define T_  512
#define I_  256
#define H_  512
#define O_  32
#define G3H (3*H_)   // 1536

// round-to-nearest-even f32 -> bf16 (bit pattern)
__device__ __forceinline__ u16 f2bf(float f){
  u32 u = __float_as_uint(f);
  u32 r = u + 0x7FFFu + ((u >> 16) & 1u);
  return (u16)(r >> 16);
}

// Load one 16x32 bf16 WMMA operand tile fragment for this lane.
// p points at element (row = lane%16 handled by caller, k = kt*32 + 8*(lane/16)).
// Per ISA layout: lane's v16bf = [k .. k+7] ++ [k+16 .. k+23]  (two 16-byte chunks).
__device__ __forceinline__ v16bf load_tile(const u16* p){
  union { v16bf bf; float4 f4[2]; } u;
  const float4* q = (const float4*)p;
  u.f4[0] = q[0];
  u.f4[1] = q[2];   // +16 elements = +32 bytes
  return u.bf;
}

#define WMMA_BF16(C, A, Bm) \
  (C) = __builtin_amdgcn_wmma_f32_16x16x32_bf16(false, (A), false, (Bm), (short)0, (C), false, false)

// ---------- kernel 1: convert f32 weights to bf16 (kept row-major) ----------
__global__ void cvt_weights(const float* __restrict__ wih, const float* __restrict__ whh,
                            u16* __restrict__ dih, u16* __restrict__ dhh){
  const int NI = G3H*I_, NH = G3H*H_;
  for (int i = blockIdx.x*blockDim.x + threadIdx.x; i < NI+NH; i += gridDim.x*blockDim.x){
    if (i < NI) dih[i]    = f2bf(wih[i]);
    else        dhh[i-NI] = f2bf(whh[i-NI]);
  }
}

// ---------- kernel 2: fused GRU scan ----------
// grid = S*2 blocks (one per sequence x 16-row half), block = 512 threads = 16 waves.
// Wave w owns output columns [32w, 32w+32) of all three gates; h stays in registers.
__global__ __launch_bounds__(512) void gru_scan(
    const float* __restrict__ x, const int* __restrict__ lengths,
    const u16* __restrict__ wih, const u16* __restrict__ whh,
    const float* __restrict__ b_ih, const float* __restrict__ b_hh,
    float* __restrict__ partial)
{
  __shared__ __align__(16) u16 h_sh[16][H_ + 8];   // bf16 hidden (A operand), padded rows
  __shared__ __align__(16) u16 x_sh[16][I_ + 8];   // bf16 x_t tile

  const int tid  = threadIdx.x;
  const int w    = tid >> 5;            // wave id 0..15
  const int lane = tid & 31;
  const int m16  = lane & 15;
  const int hi   = lane >> 4;
  const int off8 = 8*hi;

  const int bid   = blockIdx.x;         // s*2 + bhalf
  const int s     = bid >> 1;
  const int bhalf = bid & 1;

  int len = lengths[s];
  if (len > T_) len = T_;
  if (len < 0)  len = 0;

  const int cb   = w*32;
  const int col0 = cb + m16;
  const int col1 = col0 + 16;

  // per-lane bias scalars (fixed columns across all timesteps)
  const float biR0 = b_ih[col0],       bhR0 = b_hh[col0];
  const float biR1 = b_ih[col1],       bhR1 = b_hh[col1];
  const float biZ0 = b_ih[H_+col0],    bhZ0 = b_hh[H_+col0];
  const float biZ1 = b_ih[H_+col1],    bhZ1 = b_hh[H_+col1];
  const float biN0 = b_ih[2*H_+col0],  bhN0 = b_hh[2*H_+col0];
  const float biN1 = b_ih[2*H_+col1],  bhN1 = b_hh[2*H_+col1];

  // B-operand row offsets (elements) into the bf16 weight matrices
  const int ihR0 = col0*I_,        ihR1 = col1*I_;
  const int ihZ0 = (H_+col0)*I_,   ihZ1 = (H_+col1)*I_;
  const int ihN0 = (2*H_+col0)*I_, ihN1 = (2*H_+col1)*I_;
  const int hhR0 = col0*H_,        hhR1 = col1*H_;
  const int hhZ0 = (H_+col0)*H_,   hhZ1 = (H_+col1)*H_;
  const int hhN0 = (2*H_+col0)*H_, hhN1 = (2*H_+col1)*H_;

  // h0 = 0
  for (int i = tid; i < 16*(H_+8); i += 512) (&h_sh[0][0])[i] = 0;

  v8f hA = {0.f,0.f,0.f,0.f,0.f,0.f,0.f,0.f};   // h for cols col0-group (f32, persistent)
  v8f hB = {0.f,0.f,0.f,0.f,0.f,0.f,0.f,0.f};   // h for cols col1-group

  // this wave cooperatively loads x row (batch row == wave id)
  const float* xbase = x + ((size_t)(s*B_ + bhalf*16 + w))*((size_t)T_*I_);

  for (int t = 0; t < len; ++t){
    // ---- phase a: load x_t (f32) -> bf16 LDS; wave w loads row w, lane covers 8 elems
    {
      const float* px = xbase + (size_t)t*I_ + lane*8;
      float4 a0 = ((const float4*)px)[0];
      float4 a1 = ((const float4*)px)[1];
      u32 p0 = (u32)f2bf(a0.x) | ((u32)f2bf(a0.y) << 16);
      u32 p1 = (u32)f2bf(a0.z) | ((u32)f2bf(a0.w) << 16);
      u32 p2 = (u32)f2bf(a1.x) | ((u32)f2bf(a1.y) << 16);
      u32 p3 = (u32)f2bf(a1.z) | ((u32)f2bf(a1.w) << 16);
      *(uint4*)&x_sh[w][lane*8] = make_uint4(p0, p1, p2, p3);
    }
    __syncthreads();   // x_t visible; previous step's h writes visible

    v8f r0 = {0.f,0.f,0.f,0.f,0.f,0.f,0.f,0.f};
    v8f r1 = {0.f,0.f,0.f,0.f,0.f,0.f,0.f,0.f};
    v8f z0 = {0.f,0.f,0.f,0.f,0.f,0.f,0.f,0.f};
    v8f z1 = {0.f,0.f,0.f,0.f,0.f,0.f,0.f,0.f};
    v8f xn0 = {0.f,0.f,0.f,0.f,0.f,0.f,0.f,0.f};
    v8f xn1 = {0.f,0.f,0.f,0.f,0.f,0.f,0.f,0.f};
    v8f hn0 = {0.f,0.f,0.f,0.f,0.f,0.f,0.f,0.f};
    v8f hn1 = {0.f,0.f,0.f,0.f,0.f,0.f,0.f,0.f};

    // ---- phase c1: x projection, K = I_ (r,z accumulate combined; n kept separate)
    #pragma unroll 2
    for (int kt = 0; kt < I_/32; ++kt){
      const int ko = kt*32 + off8;
      v16bf A = load_tile(&x_sh[m16][ko]);
      WMMA_BF16(r0,  A, load_tile(wih + ihR0 + ko));
      WMMA_BF16(r1,  A, load_tile(wih + ihR1 + ko));
      WMMA_BF16(z0,  A, load_tile(wih + ihZ0 + ko));
      WMMA_BF16(z1,  A, load_tile(wih + ihZ1 + ko));
      WMMA_BF16(xn0, A, load_tile(wih + ihN0 + ko));
      WMMA_BF16(xn1, A, load_tile(wih + ihN1 + ko));
    }
    // ---- phase c2: h projection, K = H_
    #pragma unroll 2
    for (int kt = 0; kt < H_/32; ++kt){
      const int ko = kt*32 + off8;
      v16bf A = load_tile(&h_sh[m16][ko]);
      WMMA_BF16(r0,  A, load_tile(whh + hhR0 + ko));
      WMMA_BF16(r1,  A, load_tile(whh + hhR1 + ko));
      WMMA_BF16(z0,  A, load_tile(whh + hhZ0 + ko));
      WMMA_BF16(z1,  A, load_tile(whh + hhZ1 + ko));
      WMMA_BF16(hn0, A, load_tile(whh + hhN0 + ko));
      WMMA_BF16(hn1, A, load_tile(whh + hhN1 + ko));
    }
    __syncthreads();   // all waves done reading h_sh before we overwrite it

    // ---- phase e: GRU nonlinearity fully in registers; refresh bf16 h in LDS
    #pragma unroll
    for (int v = 0; v < 8; ++v){
      float rr = 1.f/(1.f + __expf(-(r0[v] + biR0 + bhR0)));
      float zz = 1.f/(1.f + __expf(-(z0[v] + biZ0 + bhZ0)));
      float nn = tanhf(xn0[v] + biN0 + rr*(hn0[v] + bhN0));
      hA[v] = (1.f - zz)*nn + zz*hA[v];
      h_sh[v + 8*hi][col0] = f2bf(hA[v]);

      rr = 1.f/(1.f + __expf(-(r1[v] + biR1 + bhR1)));
      zz = 1.f/(1.f + __expf(-(z1[v] + biZ1 + bhZ1)));
      nn = tanhf(xn1[v] + biN1 + rr*(hn1[v] + bhN1));
      hB[v] = (1.f - zz)*nn + zz*hB[v];
      h_sh[v + 8*hi][col1] = f2bf(hB[v]);
    }
  }

  // ---- column sums over this block's 16 rows (for batch mean)
  float s0 = 0.f, s1 = 0.f;
  #pragma unroll
  for (int v = 0; v < 8; ++v){ s0 += hA[v]; s1 += hB[v]; }
  s0 += __shfl_xor(s0, 16, 32);   // add rows 8..15 half (wave32)
  s1 += __shfl_xor(s1, 16, 32);
  if (lane < 16){
    partial[(size_t)bid*H_ + col0] = s0;
    partial[(size_t)bid*H_ + col1] = s1;
  }
}

// ---------- kernel 3: out = (mean_b h) @ W_out^T + b_out ----------
__global__ void out_proj(const float* __restrict__ partial, const float* __restrict__ W_out,
                         const float* __restrict__ b_out, float* __restrict__ out){
  int tid = blockIdx.x*blockDim.x + threadIdx.x;
  if (tid >= S_*O_) return;
  int s = tid / O_, o = tid % O_;
  const float* p0 = partial + (size_t)(2*s)*H_;
  const float* p1 = partial + (size_t)(2*s+1)*H_;
  const float* wr = W_out + (size_t)o*H_;
  float acc = 0.f;
  for (int h = 0; h < H_; ++h) acc += (p0[h] + p1[h])*wr[h];
  out[tid] = acc*(1.0f/B_) + b_out[o];
}

extern "C" void kernel_launch(void* const* d_in, const int* in_sizes, int n_in,
                              void* d_out, int out_size, void* d_ws, size_t ws_size,
                              hipStream_t stream){
  const float* x     = (const float*)d_in[0];
  const int*   lens  = (const int*)  d_in[1];
  const float* W_ih  = (const float*)d_in[2];
  const float* W_hh  = (const float*)d_in[3];
  const float* b_ih  = (const float*)d_in[4];
  const float* b_hh  = (const float*)d_in[5];
  const float* W_out = (const float*)d_in[6];
  const float* b_out = (const float*)d_in[7];
  float* out = (float*)d_out;

  // workspace layout: bf16 W_ih | bf16 W_hh | f32 partial sums [S*2][H]
  u16* wih_bf = (u16*)d_ws;
  u16* whh_bf = wih_bf + (size_t)G3H*I_;
  float* partial = (float*)(whh_bf + (size_t)G3H*H_);

  cvt_weights<<<512, 256, 0, stream>>>(W_ih, W_hh, wih_bf, whh_bf);
  gru_scan<<<S_*2, 512, 0, stream>>>(x, lens, wih_bf, whh_bf, b_ih, b_hh, partial);
  out_proj<<<1, 256, 0, stream>>>(partial, W_out, b_out, out);
}
